// Block_31636729102537
// MI455X (gfx1250) — matrix operand order repaired and checked by reading, not compiled, not run on hardware
//
#include <hip/hip_runtime.h>
#include <stdint.h>

// ---------------------------------------------------------------------------
// Transformer block forward for MI455X (gfx1250), wave32 + WMMA bf16 + TDM.
// B=8, S=1024, H=768, NH=12, HD=64, I=3072.
// ---------------------------------------------------------------------------

typedef __attribute__((ext_vector_type(16))) __bf16   v16bf;
typedef __attribute__((ext_vector_type(8)))  float    v8f;
typedef __attribute__((ext_vector_type(4)))  unsigned int u32x4;
typedef __attribute__((ext_vector_type(8)))  int      i32x8;
typedef __attribute__((ext_vector_type(4)))  int      i32x4;

#define H_DIM   768
#define S_LEN   1024
#define B_SZ    8
#define NHEADS  12
#define HD      64
#define I_DIM   3072
#define NROWS   (B_SZ * S_LEN)   /* 8192 */
#define QKV_LD  (3 * H_DIM)      /* 2304 */

#if defined(__has_builtin)
#if __has_builtin(__builtin_amdgcn_tensor_load_to_lds)
#define HAVE_TDM 1
#endif
#endif
#ifndef HAVE_TDM
#define HAVE_TDM 0
#endif

// float -> bf16 (round to nearest even)
static __device__ __forceinline__ unsigned short f2bf(float f) {
  union { float f; unsigned int u; } c; c.f = f;
  unsigned int r = (c.u + 0x7FFFu + ((c.u >> 16) & 1u)) >> 16;
  return (unsigned short)r;
}

union V16 {
  u32x4          u[2];
  unsigned short s[16];
  v16bf          v;
};

// A-matrix 16x32 bf16 layout (ISA 7.12.2): lane = M row (lane&15);
// lanes 0-15 hold K = k0+0..7 and k0+16..23; lanes 16-31 hold K = k0+8..15, k0+24..31.
static __device__ __forceinline__ v16bf load_a_tile(const unsigned short* base,
                                                    int row, int ldk, int k0, int lane) {
  int half = lane >> 4;
  const unsigned short* p = base + (long long)row * ldk + k0 + 8 * half;
  V16 t;
  t.u[0] = *reinterpret_cast<const u32x4*>(p);
  t.u[1] = *reinterpret_cast<const u32x4*>(p + 16);
  return t.v;
}

// B-matrix 32x16 bf16 layout: lane = N column (lane&15);
// lanes 0-15 hold K = k0+0..7 and k0+16..23; lanes 16-31 hold K = k0+8..15, k0+24..31.
static __device__ __forceinline__ v16bf load_b_tile(const unsigned short* base,
                                                    int col, int ldk, int k0, int lane) {
  int half = lane >> 4;
  const unsigned short* p = base + (long long)col * ldk + k0 + 8 * half;
  V16 t;
  t.u[0] = *reinterpret_cast<const u32x4*>(p);
  t.u[1] = *reinterpret_cast<const u32x4*>(p + 16);
  return t.v;
}

// LDS variants (tile rows of 32 bf16, row-major as written by TDM)
static __device__ __forceinline__ v16bf lds_a_tile(const unsigned short* tile,
                                                   int row, int lane) {
  int half = lane >> 4;
  const unsigned short* p = tile + row * 32 + 8 * half;
  V16 t;
  t.u[0] = *reinterpret_cast<const u32x4*>(p);
  t.u[1] = *reinterpret_cast<const u32x4*>(p + 16);
  return t.v;
}
static __device__ __forceinline__ v16bf lds_b_tile(const unsigned short* tile,
                                                   int col, int lane) {
  int half = lane >> 4;
  const unsigned short* p = tile + col * 32 + 8 * half;
  V16 t;
  t.u[0] = *reinterpret_cast<const u32x4*>(p);
  t.u[1] = *reinterpret_cast<const u32x4*>(p + 16);
  return t.v;
}

static __device__ __forceinline__ v8f wmma_bf16(v16bf a, v16bf b, v8f c) {
  return __builtin_amdgcn_wmma_f32_16x16x32_bf16(false, a, false, b, (short)0, c,
                                                 false, false);
}

#if HAVE_TDM
// Issue a TDM 2D tile load: Global (bf16, row stride = stride_d0 elements)
// -> LDS at byte offset lds_off. Tile = tile_d0 (elems/row) x tile_d1 (rows),
// packed contiguously row-major in LDS. D# packing per CDNA5 ISA §8.3/8.4.
static __device__ __forceinline__ void tdm_load_2d(
    unsigned lds_off, const unsigned short* gptr,
    unsigned tensor_d0, unsigned tensor_d1, unsigned stride_d0,
    unsigned tile_d0, unsigned tile_d1) {
  unsigned long long ga = (unsigned long long)(uintptr_t)gptr;
  u32x4 g0;
  g0[0] = 1u;                                        // count=1, user mode
  g0[1] = lds_off;                                   // lds_addr [63:32]
  g0[2] = (unsigned)(ga & 0xFFFFFFFFu);              // global_addr [95:64]
  g0[3] = (unsigned)((ga >> 32) & 0x01FFFFFFu)       // global_addr [120:96]
          | (2u << 30);                              // type=2 ("image") [127:126]
  i32x8 g1;
  g1[0] = (int)(1u << 16);                           // data_size=1 (2 bytes) [17:16]
  g1[1] = (int)((tensor_d0 & 0xFFFFu) << 16);        // tensor_dim0[15:0] @ [79:64... dw1 31:16]
  g1[2] = (int)(((tensor_d0 >> 16) & 0xFFFFu) |      // tensor_dim0[31:16]
                ((tensor_d1 & 0xFFFFu) << 16));      // tensor_dim1[15:0]
  g1[3] = (int)(((tensor_d1 >> 16) & 0xFFFFu) |      // tensor_dim1[31:16]
                ((tile_d0 & 0xFFFFu) << 16));        // tile_dim0 [127:112]
  g1[4] = (int)(tile_d1 & 0xFFFFu);                  // tile_dim1 [143:128], tile_dim2=0
  g1[5] = (int)stride_d0;                            // tensor_dim0_stride[31:0]
  g1[6] = 0;                                         // stride[47:32]=0, dim1_stride lo=0
  g1[7] = 0;
  i32x4 z4 = {};
#if __clang_major__ >= 23
  i32x8 z8 = {};
  __builtin_amdgcn_tensor_load_to_lds(g0, g1, z4, z4, z8, 0);
#else
  __builtin_amdgcn_tensor_load_to_lds(g0, g1, z4, z4, 0);
#endif
}
#endif

// ---------------------------------------------------------------------------
// Weight convert + transpose: w[K][N] f32 -> wT[N][K] bf16 (K-major for B tiles)
// ---------------------------------------------------------------------------
__global__ void cvt_wT_kernel(const float* __restrict__ w,
                              unsigned short* __restrict__ wT, int K, int N) {
  long long idx = (long long)blockIdx.x * blockDim.x + threadIdx.x;
  long long total = (long long)K * N;
  if (idx >= total) return;
  int n = (int)(idx / K);
  int k = (int)(idx % K);
  wT[idx] = f2bf(w[(long long)k * N + n]);
}

// ---------------------------------------------------------------------------
// LayerNorm (biased var) + cast to bf16, one row (768) per block of 256.
// ---------------------------------------------------------------------------
__global__ void __launch_bounds__(256) ln_bf16_kernel(
    const float* __restrict__ x, const float* __restrict__ g,
    const float* __restrict__ beta, unsigned short* __restrict__ out) {
  __shared__ float s_sum[256];
  __shared__ float s_sq[256];
  int row = blockIdx.x;
  int t = threadIdx.x;
  const float* xr = x + (long long)row * H_DIM;
  float v0 = xr[t], v1 = xr[t + 256], v2 = xr[t + 512];
  s_sum[t] = v0 + v1 + v2;
  s_sq[t]  = v0 * v0 + v1 * v1 + v2 * v2;
  __syncthreads();
  for (int off = 128; off > 0; off >>= 1) {
    if (t < off) { s_sum[t] += s_sum[t + off]; s_sq[t] += s_sq[t + off]; }
    __syncthreads();
  }
  float mean = s_sum[0] * (1.0f / H_DIM);
  float var  = s_sq[0] * (1.0f / H_DIM) - mean * mean;
  float rstd = rsqrtf(var + 1e-5f);
  unsigned short* orow = out + (long long)row * H_DIM;
  for (int i = t; i < H_DIM; i += 256) {
    orow[i] = f2bf((xr[i] - mean) * rstd * g[i] + beta[i]);
  }
}

// ---------------------------------------------------------------------------
// bf16 WMMA GEMM: C[M,N] = A[M,K] * WT[N,K]^T + bias, fused epilogue.
// Block = 256 threads = 8 waves (4 M x 2 N); wave tile = 32x32 (4 WMMA acc).
// TDM path: wave 0 DMAs A (32x128) and B (32x64) K-step tiles into
// double-buffered LDS (tensor_load_to_lds, TENSORcnt), overlapped with WMMA.
// mode 0: write bf16(val)            (QKV)
// mode 1: write f32(val + resid)     (proj / fc2, residual add)
// mode 2: write bf16(gelu(val))      (fc1)
// ---------------------------------------------------------------------------
__global__ void __launch_bounds__(256) gemm_bf16_kernel(
    const unsigned short* __restrict__ A, const unsigned short* __restrict__ WT,
    const float* __restrict__ bias, const float* __restrict__ resid,
    float* __restrict__ outF, unsigned short* __restrict__ outB,
    int M, int N, int K, int mode) {
  int lane = threadIdx.x & 31;
  int wid  = threadIdx.x >> 5;
  int wm   = wid >> 1;
  int wn   = wid & 1;
  int rowblk  = blockIdx.x * 128;
  int colblk  = blockIdx.y * 64;
  int rowbase = rowblk + wm * 32;
  int colbase = colblk + wn * 32;
  int half = lane >> 4;
  int l15  = lane & 15;

  v8f acc[2][2] = {};

#if HAVE_TDM
  __shared__ unsigned short sA[2][128 * 32];   // [buf][m][k]
  __shared__ unsigned short sB[2][64 * 32];    // [buf][n][k]
  if (wid == 0) {
    tdm_load_2d((unsigned)(uintptr_t)&sA[0][0], A + (size_t)rowblk * K,
                (unsigned)K, (unsigned)M, (unsigned)K, 32u, 128u);
    tdm_load_2d((unsigned)(uintptr_t)&sB[0][0], WT + (size_t)colblk * K,
                (unsigned)K, (unsigned)N, (unsigned)K, 32u, 64u);
  }
  int cur = 0;
  for (int k0 = 0; k0 < K; k0 += 32) {
    if (wid == 0) {
      __builtin_amdgcn_s_wait_tensorcnt(0);    // buf[cur] resident
    }
    __syncthreads();
    if (wid == 0 && (k0 + 32) < K) {           // prefetch next K-step while computing
      int nxt = cur ^ 1;
      tdm_load_2d((unsigned)(uintptr_t)&sA[nxt][0],
                  A + (size_t)rowblk * K + (k0 + 32),
                  (unsigned)K, (unsigned)M, (unsigned)K, 32u, 128u);
      tdm_load_2d((unsigned)(uintptr_t)&sB[nxt][0],
                  WT + (size_t)colblk * K + (k0 + 32),
                  (unsigned)K, (unsigned)N, (unsigned)K, 32u, 64u);
    }
    v16bf a0 = lds_a_tile(&sA[cur][0], wm * 32 + l15,      lane);
    v16bf a1 = lds_a_tile(&sA[cur][0], wm * 32 + 16 + l15, lane);
    v16bf b0 = lds_b_tile(&sB[cur][0], wn * 32 + l15,      lane);
    v16bf b1 = lds_b_tile(&sB[cur][0], wn * 32 + 16 + l15, lane);
    acc[0][0] = wmma_bf16(a0, b0, acc[0][0]);
    acc[0][1] = wmma_bf16(a0, b1, acc[0][1]);
    acc[1][0] = wmma_bf16(a1, b0, acc[1][0]);
    acc[1][1] = wmma_bf16(a1, b1, acc[1][1]);
    __syncthreads();                           // reads done before buf reuse
    cur ^= 1;
  }
#else
  for (int k0 = 0; k0 < K; k0 += 32) {
    v16bf a0 = load_a_tile(A, rowbase + l15,      K, k0, lane);
    v16bf a1 = load_a_tile(A, rowbase + 16 + l15, K, k0, lane);
    v16bf b0 = load_b_tile(WT, colbase + l15,      K, k0, lane);
    v16bf b1 = load_b_tile(WT, colbase + 16 + l15, K, k0, lane);
    acc[0][0] = wmma_bf16(a0, b0, acc[0][0]);
    acc[0][1] = wmma_bf16(a0, b1, acc[0][1]);
    acc[1][0] = wmma_bf16(a1, b0, acc[1][0]);
    acc[1][1] = wmma_bf16(a1, b1, acc[1][1]);
  }
#endif

  for (int tm = 0; tm < 2; ++tm) {
    for (int tn = 0; tn < 2; ++tn) {
      for (int v = 0; v < 8; ++v) {
        int grow = rowbase + tm * 16 + v + 8 * half;   // C layout: row = vgpr + 8*(lane>=16)
        int gcol = colbase + tn * 16 + l15;            //           col = lane & 15
        long long o = (long long)grow * N + gcol;
        float val = acc[tm][tn][v] + bias[gcol];
        if (mode == 0) {
          outB[o] = f2bf(val);
        } else if (mode == 1) {
          outF[o] = val + resid[o];
        } else {
          float xg = val;
          float gl = 0.5f * xg *
                     (1.0f + tanhf(0.7978845608028654f * (xg + 0.044715f * xg * xg * xg)));
          outB[o] = f2bf(gl);
        }
      }
    }
  }
}

// ---------------------------------------------------------------------------
// V transpose: qkv_bf [B*S, 2304] -> vT [B*NH, HD, S] (bf16 passthrough)
// so P@V B-tiles are per-lane contiguous along the key dimension.
// ---------------------------------------------------------------------------
__global__ void vtrans_kernel(const unsigned short* __restrict__ qkv,
                              unsigned short* __restrict__ vT) {
  int idx = blockIdx.x * blockDim.x + threadIdx.x;   // bh*65536 + d*1024 + s
  int s  = idx & (S_LEN - 1);
  int d  = (idx >> 10) & (HD - 1);
  int bh = idx >> 16;
  int b = bh / NHEADS, h = bh % NHEADS;
  vT[(long long)idx] =
      qkv[((long long)(b * S_LEN + s)) * QKV_LD + 2 * H_DIM + h * HD + d];
}

// ---------------------------------------------------------------------------
// Flash attention: one wave = 16 queries, block = 4 waves = 64 queries.
// grid = B*NH * (S/64). Online softmax; scores and P@V via WMMA bf16.
// ---------------------------------------------------------------------------
__global__ void __launch_bounds__(128) attn_kernel(
    const unsigned short* __restrict__ qkv, const unsigned short* __restrict__ vT,
    unsigned short* __restrict__ outB) {
  __shared__ float pbuf[4][16][33];   // per-wave P transpose staging
  int lane = threadIdx.x & 31;
  int wid  = threadIdx.x >> 5;
  int half = lane >> 4;
  int l15  = lane & 15;
  int bh = blockIdx.x >> 4;           // / (S/64)
  int qb = blockIdx.x & 15;
  int b = bh / NHEADS, h = bh % NHEADS;
  int q0 = qb * 64 + wid * 16;
  int qcol = h * HD;
  int kcol = H_DIM + h * HD;

  // Q tile (16 x 64) as two A tiles (K = 0..31, 32..63), held for whole loop
  int qrow = b * S_LEN + q0 + l15;
  v16bf qa0 = load_a_tile(qkv, qrow, QKV_LD, qcol + 0,  lane);
  v16bf qa1 = load_a_tile(qkv, qrow, QKV_LD, qcol + 32, lane);

  float m[8], l[8];
  for (int r = 0; r < 8; ++r) { m[r] = -3.0e38f; l[r] = 0.0f; }
  v8f o[4] = {};
  const float scale = 0.125f;         // 1/sqrt(64)

  for (int j0 = 0; j0 < S_LEN; j0 += 32) {
    // --- scores for 32 keys: two 16x16 tiles, each accumulated over d=0..63 ---
    v8f s0 = {}, s1 = {};
    {
      int krow0 = b * S_LEN + j0 + l15;
      int krow1 = krow0 + 16;
      v16bf kb00 = load_b_tile(qkv, krow0, QKV_LD, kcol + 0,  lane);
      v16bf kb01 = load_b_tile(qkv, krow0, QKV_LD, kcol + 32, lane);
      v16bf kb10 = load_b_tile(qkv, krow1, QKV_LD, kcol + 0,  lane);
      v16bf kb11 = load_b_tile(qkv, krow1, QKV_LD, kcol + 32, lane);
      s0 = wmma_bf16(qa0, kb00, s0);
      s0 = wmma_bf16(qa1, kb01, s0);
      s1 = wmma_bf16(qa0, kb10, s1);
      s1 = wmma_bf16(qa1, kb11, s1);
    }

    // --- online softmax (row = vgpr + 8*(lane>=16); 16 lanes share a row) ---
    float mx[8];
    for (int r = 0; r < 8; ++r) {
      s0[r] *= scale; s1[r] *= scale;
      mx[r] = fmaxf(s0[r], s1[r]);
    }
    for (int off = 1; off < 16; off <<= 1)
      for (int r = 0; r < 8; ++r)
        mx[r] = fmaxf(mx[r], __shfl_xor(mx[r], off, 16));

    float rs[8], alpha[8];
    for (int r = 0; r < 8; ++r) {
      float mn = fmaxf(m[r], mx[r]);
      alpha[r] = __expf(m[r] - mn);
      float p0 = __expf(s0[r] - mn);
      float p1 = __expf(s1[r] - mn);
      s0[r] = p0; s1[r] = p1;
      rs[r] = p0 + p1;
      m[r] = mn;
    }
    for (int off = 1; off < 16; off <<= 1)
      for (int r = 0; r < 8; ++r)
        rs[r] += __shfl_xor(rs[r], off, 16);
    for (int r = 0; r < 8; ++r) {
      l[r] = l[r] * alpha[r] + rs[r];
      for (int t = 0; t < 4; ++t) o[t][r] *= alpha[r];
    }

    // --- transpose P (C layout -> A layout) through LDS, cast to bf16 ---
    for (int r = 0; r < 8; ++r) {
      int prow = r + 8 * half;
      pbuf[wid][prow][l15]      = s0[r];
      pbuf[wid][prow][16 + l15] = s1[r];
    }
    asm volatile("s_wait_dscnt 0" ::: "memory");   // LDS stores visible to reads
    V16 pa;
    for (int e = 0; e < 8; ++e) {
      pa.s[e]     = f2bf(pbuf[wid][l15][8 * half + e]);
      pa.s[8 + e] = f2bf(pbuf[wid][l15][16 + 8 * half + e]);
    }

    // --- O += P @ V : four 16x16 d-tiles, keys contiguous in vT ---
    for (int t = 0; t < 4; ++t) {
      const unsigned short* vp =
          vT + ((long long)(bh * HD + 16 * t + l15)) * S_LEN + j0 + 8 * half;
      V16 vb;
      vb.u[0] = *reinterpret_cast<const u32x4*>(vp);
      vb.u[1] = *reinterpret_cast<const u32x4*>(vp + 16);
      o[t] = wmma_bf16(pa.v, vb.v, o[t]);
    }
  }

  // --- finalize: divide by softmax denom, write bf16 into [B*S, H] ---
  for (int r = 0; r < 8; ++r) {
    float inv = 1.0f / l[r];
    int qg = q0 + r + 8 * half;
    long long rb = ((long long)(b * S_LEN + qg)) * H_DIM + h * HD;
    for (int t = 0; t < 4; ++t) {
      outB[rb + 16 * t + l15] = f2bf(o[t][r] * inv);
    }
  }
}

// ---------------------------------------------------------------------------
// Host-side orchestration
// ---------------------------------------------------------------------------
extern "C" void kernel_launch(void* const* d_in, const int* in_sizes, int n_in,
                              void* d_out, int out_size, void* d_ws, size_t ws_size,
                              hipStream_t stream) {
  (void)in_sizes; (void)n_in; (void)out_size; (void)ws_size;
  const float* x      = (const float*)d_in[0];
  const float* ln1_g  = (const float*)d_in[1];
  const float* ln1_b  = (const float*)d_in[2];
  const float* qkv_w  = (const float*)d_in[3];
  const float* qkv_b  = (const float*)d_in[4];
  const float* proj_w = (const float*)d_in[5];
  const float* proj_b = (const float*)d_in[6];
  const float* ln2_g  = (const float*)d_in[7];
  const float* ln2_b  = (const float*)d_in[8];
  const float* fc1_w  = (const float*)d_in[9];
  const float* fc1_b  = (const float*)d_in[10];
  const float* fc2_w  = (const float*)d_in[11];
  const float* fc2_b  = (const float*)d_in[12];
  float* out = (float*)d_out;

  char* ws = (char*)d_ws;
  size_t off = 0;
  auto alloc = [&](size_t bytes) {
    char* p = ws + off;
    off += (bytes + 255) & ~(size_t)255;
    return p;
  };
  unsigned short* wqkvT = (unsigned short*)alloc((size_t)QKV_LD * H_DIM * 2);
  unsigned short* wprojT= (unsigned short*)alloc((size_t)H_DIM * H_DIM * 2);
  unsigned short* wfc1T = (unsigned short*)alloc((size_t)I_DIM * H_DIM * 2);
  unsigned short* wfc2T = (unsigned short*)alloc((size_t)H_DIM * I_DIM * 2);
  unsigned short* h_bf  = (unsigned short*)alloc((size_t)NROWS * H_DIM * 2);
  unsigned short* qkv_bf= (unsigned short*)alloc((size_t)NROWS * QKV_LD * 2);
  unsigned short* vT    = (unsigned short*)alloc((size_t)B_SZ * NHEADS * HD * S_LEN * 2);
  unsigned short* at_bf = (unsigned short*)alloc((size_t)NROWS * H_DIM * 2);
  float*          out1  = (float*)alloc((size_t)NROWS * H_DIM * 4);
  unsigned short* h2_bf = (unsigned short*)alloc((size_t)NROWS * H_DIM * 2);
  unsigned short* m_bf  = (unsigned short*)alloc((size_t)NROWS * I_DIM * 2);

  // 1) weights -> bf16, K-major
  {
    long long t;
    t = (long long)H_DIM * QKV_LD;
    cvt_wT_kernel<<<(t + 255) / 256, 256, 0, stream>>>(qkv_w, wqkvT, H_DIM, QKV_LD);
    t = (long long)H_DIM * H_DIM;
    cvt_wT_kernel<<<(t + 255) / 256, 256, 0, stream>>>(proj_w, wprojT, H_DIM, H_DIM);
    t = (long long)H_DIM * I_DIM;
    cvt_wT_kernel<<<(t + 255) / 256, 256, 0, stream>>>(fc1_w, wfc1T, H_DIM, I_DIM);
    t = (long long)I_DIM * H_DIM;
    cvt_wT_kernel<<<(t + 255) / 256, 256, 0, stream>>>(fc2_w, wfc2T, I_DIM, H_DIM);
  }

  // 2) LN1 -> h_bf
  ln_bf16_kernel<<<NROWS, 256, 0, stream>>>(x, ln1_g, ln1_b, h_bf);

  // 3) QKV GEMM: qkv_bf = bf16(h @ qkv_w + b)   [8192 x 2304]
  gemm_bf16_kernel<<<dim3(NROWS / 128, QKV_LD / 64), 256, 0, stream>>>(
      h_bf, wqkvT, qkv_b, nullptr, nullptr, qkv_bf, NROWS, QKV_LD, H_DIM, 0);

  // 4) V transpose for P@V
  vtrans_kernel<<<(B_SZ * NHEADS * HD * S_LEN) / 256, 256, 0, stream>>>(qkv_bf, vT);

  // 5) flash attention -> at_bf [8192 x 768]
  attn_kernel<<<B_SZ * NHEADS * (S_LEN / 64), 128, 0, stream>>>(qkv_bf, vT, at_bf);

  // 6) proj GEMM + residual: out1 = x + at @ proj_w + b
  gemm_bf16_kernel<<<dim3(NROWS / 128, H_DIM / 64), 256, 0, stream>>>(
      at_bf, wprojT, proj_b, x, out1, nullptr, NROWS, H_DIM, H_DIM, 1);

  // 7) LN2 -> h2_bf
  ln_bf16_kernel<<<NROWS, 256, 0, stream>>>(out1, ln2_g, ln2_b, h2_bf);

  // 8) FC1 GEMM + GELU -> m_bf [8192 x 3072]
  gemm_bf16_kernel<<<dim3(NROWS / 128, I_DIM / 64), 256, 0, stream>>>(
      h2_bf, wfc1T, fc1_b, nullptr, nullptr, m_bf, NROWS, I_DIM, H_DIM, 2);

  // 9) FC2 GEMM + residual -> out = out1 + m @ fc2_w + b
  gemm_bf16_kernel<<<dim3(NROWS / 128, H_DIM / 64), 256, 0, stream>>>(
      m_bf, wfc2T, fc2_b, out1, out, nullptr, NROWS, H_DIM, I_DIM, 1);
}
